// CascadeGNN_69449621176462
// MI455X (gfx1250) — matrix operand order repaired and checked by reading, not compiled, not run on hardware
//
#include <hip/hip_runtime.h>
#include <hip/hip_bf16.h>
#include <math.h>

typedef __attribute__((ext_vector_type(16))) _Float16 v16h;
typedef __attribute__((ext_vector_type(8)))  float    v8f;

#define N_NODES 50000
#define N_EDGES 1600000
#define HID 16

// ---------------- helpers ----------------

__device__ inline void atomicMaxF(float* addr, float v) {
    if (v >= 0.f) atomicMax((int*)addr, __float_as_int(v));
    else          atomicMin((unsigned int*)addr, (unsigned int)__float_as_int(v));
}

__device__ inline float gelu_exact(float v) {
    return 0.5f * v * (1.f + erff(v * 0.70710678118654752f));
}

__device__ inline float sigmoidf(float v) { return 1.f / (1.f + expf(-v)); }

__device__ inline void load8f(const float* __restrict__ p, float* t) {
    float4 a = *(const float4*)p;
    float4 b = *(const float4*)(p + 4);
    t[0] = a.x; t[1] = a.y; t[2] = a.z; t[3] = a.w;
    t[4] = b.x; t[5] = b.y; t[6] = b.z; t[7] = b.w;
}

// xor-swizzle add: single ds_swizzle_b32, imm pattern (and=0x1f, or=0, xor=M)
template <int M>
__device__ inline float swz_add(float v) {
    int r = __builtin_amdgcn_ds_swizzle(__float_as_int(v), 0x1f | (M << 10));
    return v + __int_as_float(r);
}

// ---------------- fold attention vectors through W: was = W @ a_s ------------
__global__ void k_wa(const float* __restrict__ W, const float* __restrict__ a_s,
                     const float* __restrict__ a_d, float* __restrict__ was,
                     float* __restrict__ wad, int in_dim)
{
    int k = threadIdx.x;
    if (k >= in_dim) return;
    float s1 = 0.f, s2 = 0.f;
    #pragma unroll
    for (int j = 0; j < HID; ++j) {
        float w = W[k * HID + j];
        s1 += w * a_s[j];
        s2 += w * a_d[j];
    }
    was[k] = s1; wad[k] = s2;
}

// ---------------- node transform: h = x @ W ; asrc = x@was ; adst = x@wad ----
// One wave per 16-node tile. WMMA f32_16x16x32_f16; IN_DIM compile-time.
// asrc/adst accumulated during A-fragment packing; one xor-16 swizzle combines
// the two K-halves (lane L holds K%16 in 0..7 of row L, lane L+16 holds K%16 in 8..15).
template <int IN_DIM>
__global__ void k_transform(const float* __restrict__ x, const float* __restrict__ W,
                            const float* __restrict__ was, const float* __restrict__ wad,
                            float* __restrict__ h, float* __restrict__ asrc,
                            float* __restrict__ adst, int ntiles)
{
    int wave = (blockIdx.x * blockDim.x + threadIdx.x) >> 5;
    if (wave >= ntiles) return;                 // wave-uniform: EXEC stays all-ones
    int lane = threadIdx.x & 31;
    int half = lane >> 4;
    int lrow = lane & 15;
    int row  = wave * 16 + lrow;
    const float* xr = x + (size_t)row * IN_DIM;

    v8f c = {};
    float ssrc = 0.f, sdst = 0.f;
    constexpr int NCH = (IN_DIM + 31) / 32;
    #pragma unroll
    for (int ch = 0; ch < NCH; ++ch) {
        const int kb = ch * 32;
        v16h A, B;
        // elements 0..7 : K = kb + half*8 + i
        {
            float t[8], w1[8], w2[8];
            load8f(xr + kb + half * 8, t);
            load8f(was + kb + half * 8, w1);
            load8f(wad + kb + half * 8, w2);
            #pragma unroll
            for (int i = 0; i < 8; ++i) {
                A[i] = (_Float16)t[i];
                ssrc += t[i] * w1[i];
                sdst += t[i] * w2[i];
            }
        }
        // elements 8..15: K = kb + 16 + half*8 + i (compile-time in/out of range)
        if (kb + 16 < IN_DIM) {
            float t[8], w1[8], w2[8];
            load8f(xr + kb + 16 + half * 8, t);
            load8f(was + kb + 16 + half * 8, w1);
            load8f(wad + kb + 16 + half * 8, w2);
            #pragma unroll
            for (int i = 0; i < 8; ++i) {
                A[i + 8] = (_Float16)t[i];
                ssrc += t[i] * w1[i];
                sdst += t[i] * w2[i];
            }
        } else {
            #pragma unroll
            for (int i = 0; i < 8; ++i) A[i + 8] = (_Float16)0.f;
        }
        // B: lane = column, elements 0..7: K = kb + half*8 + i, 8..15: K = kb + 16 + half*8 + i (clamped+masked)
        #pragma unroll
        for (int i = 0; i < 16; ++i) {
            int k  = kb + half * 8 + i + (i >= 8 ? 8 : 0);
            int kc = (k < IN_DIM) ? k : (IN_DIM - 1);
            float w = W[kc * HID + lrow];
            B[i] = (k < IN_DIM) ? (_Float16)w : (_Float16)0.f;
        }
        c = __builtin_amdgcn_wmma_f32_16x16x32_f16(false, A, false, B, (short)0, c,
                                                   false, false);
    }

    // combine K-halves held by lane pairs (L, L+16): one xor-16 swizzle each
    ssrc = swz_add<16>(ssrc);
    sdst = swz_add<16>(sdst);
    if (half == 0) {                 // coalesced: 16 lanes write 16 consecutive floats
        asrc[row] = ssrc;
        adst[row] = sdst;
    }

    #pragma unroll
    for (int v = 0; v < 8; ++v) {    // C layout: row = v (+8 for high half), col = lrow
        int mrow = wave * 16 + v + half * 8;
        h[(size_t)mrow * HID + lrow] = c[v];
    }
}

// ---------------- segment-softmax scaffolding ----------------

__global__ void k_init(float* __restrict__ mseg, float* __restrict__ denom,
                       float* __restrict__ num)
{
    int i = blockIdx.x * blockDim.x + threadIdx.x;
    if (i < N_NODES) { mseg[i] = __int_as_float(0xff800000); denom[i] = 0.f; }
    if (i < N_NODES * HID) num[i] = 0.f;
}

__global__ void k_edge_max(const float* __restrict__ asrc, const float* __restrict__ adst,
                           const int* __restrict__ eidx, float* __restrict__ mseg)
{
    int i = blockIdx.x * blockDim.x + threadIdx.x;
    if (i >= N_EDGES + N_NODES) return;
    int s, d;
    if (i < N_EDGES) { s = eidx[i]; d = eidx[N_EDGES + i]; }
    else             { s = d = i - N_EDGES; }               // self-loops
    float z = asrc[s] + adst[d];
    float l = z > 0.f ? z : 0.2f * z;                       // leaky_relu(0.2)
    atomicMaxF(&mseg[d], l);
}

__global__ void k_edge_sum(const float* __restrict__ asrc, const float* __restrict__ adst,
                           const float* __restrict__ h, const int* __restrict__ eidx,
                           const float* __restrict__ mseg, float* __restrict__ denom,
                           float* __restrict__ num)
{
    int i = blockIdx.x * blockDim.x + threadIdx.x;
    if (i >= N_EDGES + N_NODES) return;
    int s, d;
    if (i < N_EDGES) { s = eidx[i]; d = eidx[N_EDGES + i]; }
    else             { s = d = i - N_EDGES; }
    float z = asrc[s] + adst[d];
    float l = z > 0.f ? z : 0.2f * z;
    float e = expf(l - mseg[d]);
    atomicAdd(&denom[d], e);
    const float* hs = h + (size_t)s * HID;
    float* nd = num + (size_t)d * HID;
    #pragma unroll
    for (int j = 0; j < HID; ++j) atomicAdd(&nd[j], e * hs[j]);
}

__global__ void k_finalize(const float* __restrict__ num, const float* __restrict__ denom,
                           const float* __restrict__ b, float* __restrict__ xout)
{
    int i = blockIdx.x * blockDim.x + threadIdx.x;
    if (i >= N_NODES * HID) return;
    int node = i >> 4, j = i & 15;
    float v = num[i] / denom[node] + b[j];
    xout[i] = gelu_exact(v);
}

// ---------------- edge MLP: sigmoid(relu([xs|xd|ee] @ Wm1 + bm1) @ Wm2 + bm2) -
// One wave per 16-edge tile; K = 48 padded to 64 -> two WMMAs. Second layer is a
// 16-wide dot per row: level-wise xor-swizzle tree over all 8 rows at once.
__global__ void k_mlp(const float* __restrict__ x, const int* __restrict__ eidx,
                      const float* __restrict__ eemb,
                      const float* __restrict__ Wm1, const float* __restrict__ bm1,
                      const float* __restrict__ Wm2, const float* __restrict__ bm2,
                      float* __restrict__ out, int ntiles)
{
    int wave = (blockIdx.x * blockDim.x + threadIdx.x) >> 5;
    if (wave >= ntiles) return;
    int lane = threadIdx.x & 31;
    int half = lane >> 4;
    int lrow = lane & 15;

    // B matrices (loop-invariant): Wm1 rows 0..31 -> B0, rows 32..47 (+pad) -> B1
    v16h B0, B1;
    #pragma unroll
    for (int i = 0; i < 16; ++i) {
        int k0 = half * 8 + i + (i >= 8 ? 8 : 0);
        int k1 = 32 + k0;
        int kc = (k1 < 48) ? k1 : 47;
        float w1 = Wm1[k0 * HID + lrow];
        float w2 = Wm1[kc * HID + lrow];
        B0[i] = (_Float16)w1;
        B1[i] = (k1 < 48) ? (_Float16)w2 : (_Float16)0.f;
    }

    int e = wave * 16 + lrow;
    int s = eidx[e], d = eidx[N_EDGES + e];
    const float* xs = x    + (size_t)s * HID + half * 8;
    const float* xd = x    + (size_t)d * HID + half * 8;
    const float* ep = eemb + (size_t)e * HID + half * 8;

    // prefetch next tile's streaming edge_emb rows
    if (e + 16 < N_EDGES) __builtin_prefetch(eemb + (size_t)(e + 16) * HID, 0, 1);

    float ts[8], td[8], te[8];
    load8f(xs, ts);
    load8f(xd, td);
    load8f(ep, te);

    v16h A0, A1;                        // A0: K0-15 = x[src], K16-31 = x[dst]
    #pragma unroll                      // A1: K32-47 = edge_emb, K48-63 = 0
    for (int i = 0; i < 8; ++i) {
        A0[i]     = (_Float16)ts[i];
        A0[i + 8] = (_Float16)td[i];
        A1[i]     = (_Float16)te[i];
        A1[i + 8] = (_Float16)0.f;
    }

    v8f c = {};
    c = __builtin_amdgcn_wmma_f32_16x16x32_f16(false, A0, false, B0, (short)0, c, false, false);
    c = __builtin_amdgcn_wmma_f32_16x16x32_f16(false, A1, false, B1, (short)0, c, false, false);

    // hmid = relu(C + bm1); t[v] = hmid[row v]·Wm2 reduced level-wise over lanes
    float b1v = bm1[lrow], w2v = Wm2[lrow], b2v = bm2[0];
    float t[8];
    #pragma unroll
    for (int v = 0; v < 8; ++v) {
        float hm = c[v] + b1v;
        hm = hm > 0.f ? hm : 0.f;
        t[v] = hm * w2v;
    }
    #pragma unroll
    for (int v = 0; v < 8; ++v) t[v] = swz_add<1>(t[v]);
    #pragma unroll
    for (int v = 0; v < 8; ++v) t[v] = swz_add<2>(t[v]);
    #pragma unroll
    for (int v = 0; v < 8; ++v) t[v] = swz_add<4>(t[v]);
    #pragma unroll
    for (int v = 0; v < 8; ++v) t[v] = swz_add<8>(t[v]);

    if (lrow == 0) {                 // rows half*8 .. half*8+7 -> contiguous output
        float4 r0, r1;
        r0.x = sigmoidf(t[0] + b2v); r0.y = sigmoidf(t[1] + b2v);
        r0.z = sigmoidf(t[2] + b2v); r0.w = sigmoidf(t[3] + b2v);
        r1.x = sigmoidf(t[4] + b2v); r1.y = sigmoidf(t[5] + b2v);
        r1.z = sigmoidf(t[6] + b2v); r1.w = sigmoidf(t[7] + b2v);
        float* op = out + wave * 16 + half * 8;
        *(float4*)op       = r0;
        *(float4*)(op + 4) = r1;
    }
}

// ---------------- launcher ----------------

extern "C" void kernel_launch(void* const* d_in, const int* in_sizes, int n_in,
                              void* d_out, int out_size, void* d_ws, size_t ws_size,
                              hipStream_t stream)
{
    (void)in_sizes; (void)n_in; (void)out_size; (void)ws_size;
    const int*   edge_index = (const int*)  d_in[0];
    const float* node_emb   = (const float*)d_in[1];
    const float* edge_emb   = (const float*)d_in[2];
    const float* W1   = (const float*)d_in[3];
    const float* as1  = (const float*)d_in[4];
    const float* ad1  = (const float*)d_in[5];
    const float* b1   = (const float*)d_in[6];
    const float* W2   = (const float*)d_in[7];
    const float* as2  = (const float*)d_in[8];
    const float* ad2  = (const float*)d_in[9];
    const float* b2   = (const float*)d_in[10];
    const float* Wm1  = (const float*)d_in[11];
    const float* bm1  = (const float*)d_in[12];
    const float* Wm2  = (const float*)d_in[13];
    const float* bm2  = (const float*)d_in[14];
    float* out = (float*)d_out;

    // workspace layout (floats)
    float* h     = (float*)d_ws;              // N*16
    float* num   = h    + N_NODES * HID;      // N*16
    float* xA    = num  + N_NODES * HID;      // N*16
    float* xB    = xA   + N_NODES * HID;      // N*16
    float* asrc  = xB   + N_NODES * HID;      // N
    float* adst  = asrc + N_NODES;            // N
    float* mseg  = adst + N_NODES;            // N
    float* denom = mseg + N_NODES;            // N
    float* was   = denom + N_NODES;           // 64
    float* wad   = was + 64;                  // 64

    const int ntiles_n = N_NODES / 16;        // 3125 (exact)
    const int ntiles_e = N_EDGES / 16;        // 100000 (exact)
    dim3 blk(256);                            // 8 wave32 per block
    int tb  = (ntiles_n + 7) / 8;
    int nb  = (N_NODES * HID + 255) / 256;
    int eb  = (N_EDGES + N_NODES + 255) / 256;
    int mb  = (ntiles_e + 7) / 8;

    // ---- GAT layer 1 (in_dim = 64) ----
    k_wa           <<<1, 64, 0, stream>>>(W1, as1, ad1, was, wad, 64);
    k_transform<64><<<tb, blk, 0, stream>>>(node_emb, W1, was, wad, h, asrc, adst, ntiles_n);
    k_init         <<<nb, blk, 0, stream>>>(mseg, denom, num);
    k_edge_max     <<<eb, blk, 0, stream>>>(asrc, adst, edge_index, mseg);
    k_edge_sum     <<<eb, blk, 0, stream>>>(asrc, adst, h, edge_index, mseg, denom, num);
    k_finalize     <<<nb, blk, 0, stream>>>(num, denom, b1, xA);

    // ---- GAT layer 2 (in_dim = 16) ----
    k_wa           <<<1, 64, 0, stream>>>(W2, as2, ad2, was, wad, 16);
    k_transform<16><<<tb, blk, 0, stream>>>(xA, W2, was, wad, h, asrc, adst, ntiles_n);
    k_init         <<<nb, blk, 0, stream>>>(mseg, denom, num);
    k_edge_max     <<<eb, blk, 0, stream>>>(asrc, adst, edge_index, mseg);
    k_edge_sum     <<<eb, blk, 0, stream>>>(asrc, adst, h, edge_index, mseg, denom, num);
    k_finalize     <<<nb, blk, 0, stream>>>(num, denom, b2, xB);

    // ---- edge MLP ----
    k_mlp<<<mb, blk, 0, stream>>>(xB, edge_index, edge_emb, Wm1, bm1, Wm2, bm2, out, ntiles_e);
}
